// GaussianUpsampling_6975026888888
// MI455X (gfx1250) — matrix operand order repaired and checked by reading, not compiled, not run on hardware
//
#include <hip/hip_runtime.h>

typedef __attribute__((ext_vector_type(16))) _Float16     v16h;
typedef __attribute__((ext_vector_type(8)))  float        v8f;
typedef __attribute__((ext_vector_type(4)))  unsigned int v4u;

#define NB    16      // batch
#define LTXT  512     // T_text
#define ADIM  256     // adim
#define TFEAT 4096    // T_feats
// energy in exp2 domain: e2 = -(DELTA*log2(e)) * (t-c)^2 ; p = exp2(e2 - mx2)
#define K2    0.14426950408889634f
#define CMASK 1.0e19f // masked token center: (t-c)^2 ~ 1e38 -> exp2 -> 0
// LDS slab: 128 dims x 128 tokens (f16), row stride 136 halves = 272 B
// -> 16-byte aligned rows, lane-to-lane spread of 4 banks (no ds_load_b128 conflicts)
#define LPAD  136
#define SLABH (128 * LPAD)   // halves per slab buffer

// ---------------------------------------------------------------------------
// Kernel 1: c[b,l] = cumsum(ds)[l] - ds[l]/2 ; d_mask folded in (masked -> 1e19)
// one wave32 per batch
// ---------------------------------------------------------------------------
__global__ __launch_bounds__(32) void gauss_centers_kernel(
    const int* __restrict__ ds, const unsigned char* __restrict__ d_masks,
    float* __restrict__ c)
{
    const int b    = blockIdx.x;
    const int lane = threadIdx.x;              // 0..31, 16 tokens each
    const int*           dsb = ds      + b * LTXT;
    const unsigned char* dmb = d_masks + b * LTXT;
    float*               cb  = c       + b * LTXT;

    float v[16];
    float lsum = 0.f;
#pragma unroll
    for (int i = 0; i < 16; ++i) { v[i] = (float)dsb[lane * 16 + i]; lsum += v[i]; }

    float inc = lsum;
#pragma unroll
    for (int off = 1; off < 32; off <<= 1) {
        float n = __shfl_up(inc, off, 32);
        if (lane >= off) inc += n;
    }
    float run = inc - lsum;                    // exclusive prefix
#pragma unroll
    for (int i = 0; i < 16; ++i) {
        run += v[i];
        float cv = run - 0.5f * v[i];          // cumsum - ds/2
        cb[lane * 16 + i] = dmb[lane * 16 + i] ? cv : CMASK;
    }
}

// ---------------------------------------------------------------------------
// Kernel 2: hs (B,L,D) f32 -> hsT (B,D,L) f16, LDS tile transpose (32x32)
// grid: (L/32, D/32, NB), block (32,8)
// ---------------------------------------------------------------------------
__global__ __launch_bounds__(256) void hs_transpose_f16_kernel(
    const float* __restrict__ hs, _Float16* __restrict__ hsT)
{
    __shared__ float tile[32][33];
    const int lx = threadIdx.x, ty = threadIdx.y;   // 32 x 8
    const int l0 = blockIdx.x * 32, d0 = blockIdx.y * 32, b = blockIdx.z;

    const float* src = hs + ((size_t)b * LTXT + l0) * ADIM + d0;
#pragma unroll
    for (int yy = 0; yy < 32; yy += 8)
        tile[ty + yy][lx] = src[(size_t)(ty + yy) * ADIM + lx];
    __syncthreads();

    _Float16* dst = hsT + ((size_t)b * ADIM + d0) * LTXT + l0;
#pragma unroll
    for (int yy = 0; yy < 32; yy += 8)
        dst[(size_t)(ty + yy) * LTXT + lx] = (_Float16)tile[lx][ty + yy];
}

// ---------------------------------------------------------------------------
// async global->LDS helpers (CDNA5: GLOBAL_LOAD_ASYNC_TO_LDS_B128, ASYNCcnt)
// ---------------------------------------------------------------------------
__device__ __forceinline__ void async_ld_b128(unsigned lds_off, unsigned long long gaddr)
{
    asm volatile("global_load_async_to_lds_b128 %0, %1, off"
                 :: "v"(lds_off), "v"(gaddr) : "memory");
}
#define WAIT_ASYNC(N) asm volatile("s_wait_asynccnt %0" :: "n"(N) : "memory")

// ---------------------------------------------------------------------------
// Kernel 3: fused energy -> softmax -> (p_attn @ hs) via v_wmma_f32_16x16x32_f16
// grid: (TFEAT/128, ADIM/128, NB), block: 256 threads (8 waves of 32)
// hsT slabs stream into double-buffered LDS via async-LDS loads.
// ---------------------------------------------------------------------------
__global__ __launch_bounds__(256) void gauss_upsample_wmma_kernel(
    const _Float16* __restrict__ hsT,
    const unsigned char* __restrict__ h_masks,
    const float* __restrict__ c_all,
    float* __restrict__ out)
{
    __shared__ __align__(16) _Float16 sB[2 * SLABH]; // double-buffered hs^T slab
    __shared__ float sC[LTXT];
    __shared__ float sInv[128];                      // per-row 1/softmax-sum

    const int tid  = threadIdx.x;
    const int lane = tid & 31;
    const int wave = tid >> 5;          // 0..7 -> 16 frames each
    const int m    = lane & 15;
    const int hi   = lane >> 4;         // 0/1: K-half selector per WMMA layout

    const int rowBase = blockIdx.x * 128;
    const int dBase   = blockIdx.y * 128;
    const int b       = blockIdx.z;

    const _Float16*      hsTb = hsT     + ((size_t)b * ADIM + dBase) * LTXT;
    const float*         cb   = c_all   + (size_t)b * LTXT;
    const unsigned char* hmb  = h_masks + (size_t)b * TFEAT;

    // LDS byte offsets of the two slab buffers (generic addr low 32 bits)
    const unsigned sb0 = (unsigned)(unsigned long long)(void*)&sB[0];

    // issue one slab (32 KB): 8 x 16B transfers per thread, coalesced
    auto issue_slab = [&](int kb, int buf) {
        const _Float16* g0 = hsTb + kb * 128;
        const unsigned  l0 = sb0 + (unsigned)(buf * SLABH * 2);
#pragma unroll
        for (int it = 0; it < 8; ++it) {
            int i  = tid + it * 256;    // 0..2047
            int dl = i >> 4;            // dim row 0..127
            int sg = i & 15;            // 16B segment along l
            async_ld_b128(l0 + (unsigned)(dl * (LPAD * 2) + sg * 16),
                          (unsigned long long)(g0 + (size_t)dl * LTXT + sg * 8));
        }
    };

    // stage centers (mask already folded in)
    for (int i = tid; i < LTXT; i += 256) sC[i] = cb[i];

    // prologue: start slab 0 while we do pass 1
    issue_slab(0, 0);
    __syncthreads();                    // sC visible

    // this lane's frame (row) index; masked frames use t=0 like the reference
    const int   t  = rowBase + wave * 16 + m;
    const float tt = (float)t * (hmb[t] ? 1.0f : 0.0f);

    // ---- pass 1: row max of exp2-domain energy (lane pair splits tokens) ---
    float mx = -3.0e38f;
    for (int k = hi * 256; k < hi * 256 + 256; ++k) {
        float d = tt - sC[k];
        mx = fmaxf(mx, d * (d * -K2));
    }
    mx = fmaxf(mx, __shfl_xor(mx, 16, 32));
    if (mx < -1.0e30f) mx = 0.0f;       // fully-masked row guard

    // ---- main loop: A (= exp2(e2-mx)) in-register, WMMA against LDS B ------
    v8f  acc[8] = {};
    float lsum = 0.0f;

    for (int kb = 0; kb < LTXT / 128; ++kb) {
        const int cur = kb & 1;
        if (kb + 1 < LTXT / 128) {
            issue_slab(kb + 1, cur ^ 1);   // readers of that buffer are done
            WAIT_ASYNC(8);                 // oldest 8 (= slab kb) retired
        } else {
            WAIT_ASYNC(0);
        }
        __syncthreads();                   // all waves' transfers visible

        const _Float16* slab = &sB[cur * SLABH];
        for (int kc = 0; kc < 4; ++kc) {
            const int l0g = kb * 128 + kc * 32;   // global token base of chunk
            const int l0s = kc * 32;              // in-slab token base

            // preload ALL 8 B fragments (16 ds_load_b128 in flight)
            union { v16h h; v4u u[2]; } bf[8];
#pragma unroll
            for (int nt = 0; nt < 8; ++nt) {
                const v4u* bp = (const v4u*)&slab[(nt * 16 + m) * LPAD + l0s + hi * 16];
                bf[nt].u[0] = bp[0];
                bf[nt].u[1] = bp[1];
            }

            // A fragment, 16x32 f16 layout:
            // lane(hi,m): K = hi*16+e
            v16h a;
#pragma unroll
            for (int e = 0; e < 16; ++e) {
                int kg   = l0g + hi * 16 + e;
                float d  = tt - sC[kg];
                float pv = __builtin_amdgcn_exp2f(d * (d * -K2) - mx);
                lsum += pv;
                a[e] = (_Float16)pv;
            }

            // 8 back-to-back WMMAs
#pragma unroll
            for (int nt = 0; nt < 8; ++nt) {
                acc[nt] = __builtin_amdgcn_wmma_f32_16x16x32_f16(
                    false, a, false, bf[nt].h, (short)0, acc[nt], false, false);
            }
        }
        __syncthreads();                   // all reads done before buffer reuse
    }

    // ---- softmax denominator (linear post-scale) ---------------------------
    float total = lsum + __shfl_xor(lsum, 16, 32);
    if (lane < 16) sInv[wave * 16 + m] = 1.0f / total;
    __syncthreads();

    // ---- store: C layout -> VGPR r holds row (hi*8 + r), col m -------------
    float* outb = out + ((size_t)b * TFEAT + rowBase) * ADIM + dBase;
#pragma unroll
    for (int r = 0; r < 8; ++r) {
        const int  rowloc = wave * 16 + hi * 8 + r;
        const float inv   = sInv[rowloc];
        float* orow = outb + (size_t)rowloc * ADIM;
#pragma unroll
        for (int nt = 0; nt < 8; ++nt) {
            orow[nt * 16 + m] = acc[nt][r] * inv;
        }
    }
}

// ---------------------------------------------------------------------------
extern "C" void kernel_launch(void* const* d_in, const int* in_sizes, int n_in,
                              void* d_out, int out_size, void* d_ws, size_t ws_size,
                              hipStream_t stream)
{
    const float*         hs = (const float*)d_in[0];
    const int*           ds = (const int*)d_in[1];
    const unsigned char* hm = (const unsigned char*)d_in[2];
    const unsigned char* dm = (const unsigned char*)d_in[3];
    float* out = (float*)d_out;

    float*    c   = (float*)d_ws;                       // 32 KB
    _Float16* hsT = (_Float16*)((char*)d_ws + NB * LTXT * sizeof(float)); // 4 MB

    gauss_centers_kernel<<<dim3(NB), dim3(32), 0, stream>>>(ds, dm, c);

    hs_transpose_f16_kernel<<<dim3(LTXT / 32, ADIM / 32, NB), dim3(32, 8), 0, stream>>>(hs, hsT);

    dim3 grid(TFEAT / 128, ADIM / 128, NB);             // 32 x 2 x 16 = 1024 blocks
    gauss_upsample_wmma_kernel<<<grid, dim3(256), 0, stream>>>(hsT, hm, c, out);
}
